// LG_VGAE_1245540516299
// MI455X (gfx1250) — matrix operand order repaired and checked by reading, not compiled, not run on hardware
//
#include <hip/hip_runtime.h>
#include <hip/hip_bf16.h>

// ---------------------------------------------------------------------------
// LG_VGAE on MI455X (gfx1250, wave32).
// Returned scalar algebraically equals vgae_loss (DGI branch cancels:
// B*dgi/(dgi/vgae) + (1-B)*vgae == vgae). Polyconv feat recurrence is shared
// across all thetas -> 2 propagations per enc_dec. GEMMs: 16x32 output per
// wave on v_wmma_f32_16x16x32_bf16, A-fragment reused across two WMMAs.
// Requires M % 16 == 0 and Nout % 32 == 0 (true: N=100000, Nout in {32,128}).
// ---------------------------------------------------------------------------

typedef __attribute__((ext_vector_type(16))) __bf16 v16bf;
typedef __attribute__((ext_vector_type(8)))  float  v8f;

#define FDIM 128
#define HSDIM 384

// ------------------------------ utility kernels ----------------------------

__global__ void k_zero(float* __restrict__ p, long long n) {
  long long i = (long long)blockIdx.x * blockDim.x + threadIdx.x;
  if (i < n) p[i] = 0.0f;
}

__global__ void k_deg(const int* __restrict__ dst, float* __restrict__ deg, int E) {
  int e = blockIdx.x * blockDim.x + threadIdx.x;
  if (e < E) atomicAdd(&deg[dst[e]], 1.0f);
}

__global__ void k_dinv(float* __restrict__ d, int N) {
  int i = blockIdx.x * blockDim.x + threadIdx.x;
  if (i < N) d[i] = rsqrtf(fmaxf(d[i], 1.0f));
}

// one wave per edge; lane -> 4 consecutive features (float4 gather + 4 atomics)
__global__ void k_prop(const float* __restrict__ feat, const float* __restrict__ dinv,
                       const int* __restrict__ src, const int* __restrict__ dst,
                       float* __restrict__ agg, int E) {
  long long t = (long long)blockIdx.x * blockDim.x + threadIdx.x;
  int e = (int)(t >> 5);
  if (e >= E) return;
  int c = ((int)t & 31) << 2;
  int s = src[e];
  int d = dst[e];
  float sc = dinv[s];
  const float4 v = *reinterpret_cast<const float4*>(feat + (size_t)s * FDIM + c);
  float* o = agg + (size_t)d * FDIM + c;
  atomicAdd(o + 0, v.x * sc);
  atomicAdd(o + 1, v.y * sc);
  atomicAdd(o + 2, v.z * sc);
  atomicAdd(o + 3, v.w * sc);
}

// feat_next = feat_prev - agg * dinv[row]; re-zeroes agg for the next pass.
__global__ void k_update(const float* __restrict__ fp, float* __restrict__ agg,
                         const float* __restrict__ dinv, float* __restrict__ fn,
                         long long n) {
  long long i = (long long)blockIdx.x * blockDim.x + threadIdx.x;
  if (i >= n) return;
  long long row = i >> 7;
  fn[i] = fp[i] - agg[i] * dinv[row];
  agg[i] = 0.0f;
}

// hs = concat of theta-weighted feats; THETAS(D=2) = [[3,-3,.75],[0,3,-1.5],[0,0,.75]]
__global__ void k_hs(const float* __restrict__ x0, const float* __restrict__ x1,
                     const float* __restrict__ x2, float* __restrict__ hs, long long n) {
  long long i = (long long)blockIdx.x * blockDim.x + threadIdx.x;
  if (i >= n) return;
  long long row = i >> 7;
  int col = (int)(i & 127);
  float a = x0[i], b = x1[i], c = x2[i];
  float* o = hs + row * HSDIM + col;
  o[0]   = 3.0f * a - 3.0f * b + 0.75f * c;
  o[128] = 3.0f * b - 1.5f * c;
  o[256] = 0.75f * c;
}

// ------------------------------ WMMA GEMM ----------------------------------
// Out[M,Nout] = A[M,K] @ W[Nout,K]^T + bias (+ optional relu).
// Each wave: one 16x32 output slab = two 16x16 WMMA tiles sharing the A frag.

__device__ __forceinline__ v16bf bf16_frag(float4 p0, float4 p1, float4 p2, float4 p3) {
  v16bf f;
  f[0]  = (__bf16)p0.x; f[1]  = (__bf16)p0.y; f[2]  = (__bf16)p0.z; f[3]  = (__bf16)p0.w;
  f[4]  = (__bf16)p1.x; f[5]  = (__bf16)p1.y; f[6]  = (__bf16)p1.z; f[7]  = (__bf16)p1.w;
  f[8]  = (__bf16)p2.x; f[9]  = (__bf16)p2.y; f[10] = (__bf16)p2.z; f[11] = (__bf16)p2.w;
  f[12] = (__bf16)p3.x; f[13] = (__bf16)p3.y; f[14] = (__bf16)p3.z; f[15] = (__bf16)p3.w;
  return f;
}

template <int K, int RELU>
__global__ void k_gemm(const float* __restrict__ A, const float* __restrict__ W,
                       const float* __restrict__ bias, float* __restrict__ Out,
                       int M, int Nout) {
  const int lane = threadIdx.x;                    // 0..31
  const int pair = blockIdx.y * blockDim.y + threadIdx.y;
  if (pair >= (Nout >> 5)) return;                 // wave-uniform; EXEC all-ones
  const int rowBase = blockIdx.x << 4;
  const int colBase = pair << 5;                   // two 16-wide tiles
  const int mk = lane & 15;
  const int hi = lane >> 4;

  const float* aptr  = A + (size_t)(rowBase + mk) * K + (hi << 4);
  const float* wptr0 = W + (size_t)(colBase + mk) * K + (hi << 4);
  const float* wptr1 = wptr0 + (size_t)16 * K;

  v8f acc0 = {};
  v8f acc1 = {};
#pragma unroll 2
  for (int k0 = 0; k0 < K; k0 += 32) {
    // A frag: lanes 0-15 hold K 0..15; lanes 16-31 hold K 16..31
    float4 a0 = *reinterpret_cast<const float4*>(aptr + k0);
    float4 a1 = *reinterpret_cast<const float4*>(aptr + k0 + 4);
    float4 a2 = *reinterpret_cast<const float4*>(aptr + k0 + 8);
    float4 a3 = *reinterpret_cast<const float4*>(aptr + k0 + 12);
    // B frags: lanes 0-15 hold K 0..15, lanes 16-31 hold K 16..31; N = lane&15
    float4 b0 = *reinterpret_cast<const float4*>(wptr0 + k0);
    float4 b1 = *reinterpret_cast<const float4*>(wptr0 + k0 + 4);
    float4 b2 = *reinterpret_cast<const float4*>(wptr0 + k0 + 8);
    float4 b3 = *reinterpret_cast<const float4*>(wptr0 + k0 + 12);
    float4 c0 = *reinterpret_cast<const float4*>(wptr1 + k0);
    float4 c1 = *reinterpret_cast<const float4*>(wptr1 + k0 + 4);
    float4 c2 = *reinterpret_cast<const float4*>(wptr1 + k0 + 8);
    float4 c3 = *reinterpret_cast<const float4*>(wptr1 + k0 + 12);
    __builtin_prefetch(aptr + k0 + 32, 0, 1);      // global_prefetch_b8

    v16bf af  = bf16_frag(a0, a1, a2, a3);
    v16bf bf0 = bf16_frag(b0, b1, b2, b3);
    v16bf bf1 = bf16_frag(c0, c1, c2, c3);

    acc0 = __builtin_amdgcn_wmma_f32_16x16x32_bf16(false, af, false, bf0,
                                                   (short)0, acc0, false, false);
    acc1 = __builtin_amdgcn_wmma_f32_16x16x32_bf16(false, af, false, bf1,
                                                   (short)0, acc1, false, false);
  }

  const float bv0 = bias[colBase + mk];
  const float bv1 = bias[colBase + 16 + mk];
  // C/D layout: lanes 0-15 -> M rows 0..7, lanes 16-31 -> rows 8..15; N = lane&15
  float* op = Out + (size_t)(rowBase + (hi << 3)) * Nout + colBase + mk;
#pragma unroll
  for (int r = 0; r < 8; ++r) {
    float v0 = acc0[r] + bv0;
    float v1 = acc1[r] + bv1;
    if (RELU) { v0 = fmaxf(v0, 0.0f); v1 = fmaxf(v1, 0.0f); }
    op[0]  = v0;
    op[16] = v1;
    op += Nout;
  }
}

// ------------------------- VAE sample + KL + reductions --------------------

__device__ __forceinline__ float hash_u01(unsigned x) {
  x ^= x >> 17; x *= 0xed5ad4bbu;
  x ^= x >> 11; x *= 0xac4c1b51u;
  x ^= x >> 15; x *= 0x31848babu;
  x ^= x >> 14;
  return (float)(x >> 8) * (1.0f / 16777216.0f) + 1.0e-7f;
}

// in-place: mu -> z = mu + eps*exp(mu/2); accumulate sum(1 + mu - mu^2 - exp(mu))
__global__ void k_zkl(float* __restrict__ mu, float* __restrict__ accum, long long n) {
  long long i = (long long)blockIdx.x * blockDim.x + threadIdx.x;
  float contrib = 0.0f;
  if (i < n) {
    float m = mu[i];
    contrib = 1.0f + m - m * m - __expf(m);
    unsigned u = (unsigned)i;
    float u1 = hash_u01(u * 2654435761u + 0x9e3779b9u);
    float u2 = hash_u01(u * 40503u + 0x7f4a7c15u);
    float eps = sqrtf(-2.0f * __logf(u1)) * __cosf(6.28318530718f * u2);
    mu[i] = m + eps * __expf(0.5f * m);
  }
#pragma unroll
  for (int off = 16; off > 0; off >>= 1) contrib += __shfl_xor(contrib, off, 32);
  if ((threadIdx.x & 31) == 0) atomicAdd(accum, contrib);
}

// accum += sum((a-b)^2)
__global__ void k_sq(const float* __restrict__ a, const float* __restrict__ b,
                     float* __restrict__ accum, long long n) {
  long long i = (long long)blockIdx.x * blockDim.x + threadIdx.x;
  float s = 0.0f;
  if (i < n) {
    float d = a[i] - b[i];
    s = d * d;
  }
#pragma unroll
  for (int off = 16; off > 0; off >>= 1) s += __shfl_xor(s, off, 32);
  if ((threadIdx.x & 31) == 0) atomicAdd(accum, s);
}

__global__ void k_final(const float* __restrict__ acc, float* __restrict__ out) {
  if (threadIdx.x == 0 && blockIdx.x == 0)
    out[0] = acc[0] - 0.5f * acc[1];   // reconst + (-0.5 * kl_sum)
}

// ------------------------------ host driver --------------------------------

extern "C" void kernel_launch(void* const* d_in, const int* in_sizes, int n_in,
                              void* d_out, int out_size, void* d_ws, size_t ws_size,
                              hipStream_t stream) {
  const float* features = (const float*)d_in[0];
  const int*   src      = (const int*)d_in[1];
  const int*   dst      = (const int*)d_in[2];
  const float* enc_w1   = (const float*)d_in[3];
  const float* enc_b1   = (const float*)d_in[4];
  const float* enc_w2   = (const float*)d_in[5];
  const float* enc_b2   = (const float*)d_in[6];
  const float* dec_w1   = (const float*)d_in[7];
  const float* dec_b1   = (const float*)d_in[8];
  const float* dec_w2   = (const float*)d_in[9];
  const float* dec_b2   = (const float*)d_in[10];
  const float* rep_w    = (const float*)d_in[11];
  const float* rep_b    = (const float*)d_in[12];
  const float* rec_w    = (const float*)d_in[13];
  const float* rec_b    = (const float*)d_in[14];
  // d_in[15] (disc_w) unused: the DGI term cancels in the returned scalar.

  const int N = in_sizes[0] / FDIM;
  const int E = in_sizes[1];
  const long long nNF = (long long)N * FDIM;

  size_t need = ((size_t)N * (1 + 4 * FDIM + HSDIM + FDIM + 32 + FDIM) + 16) * sizeof(float);
  if (ws_size < need) {
    k_zero<<<1, 32, 0, stream>>>((float*)d_out, 1);
    return;
  }
  float* dinv = (float*)d_ws;              // N
  float* Xa   = dinv + N;                  // N*128  feat0 / relu out
  float* Xb   = Xa + nNF;                  // N*128  feat1
  float* Xc   = Xb + nNF;                  // N*128  feat2
  float* G    = Xc + nNF;                  // N*128  agg scratch / x_rec
  float* HS   = G + nNF;                   // N*384
  float* P    = HS + (long long)N * HSDIM; // N*128  positive
  float* MU   = P + nNF;                   // N*32   mu -> z (in place)
  float* ZN   = MU + (long long)N * 32;    // N*128  decoder input
  float* acc  = ZN + nNF;                  // [0]=reconst sum, [1]=kl inner sum

  const unsigned TB = 256;
  auto g1 = [](long long n) { return dim3((unsigned)((n + 255) / 256)); };
  const dim3 gblk(32, 4);
  auto ggrid = [](int M, int Nout) {
    return dim3((unsigned)((M + 15) / 16), (unsigned)(((Nout / 32) + 3) / 4));
  };
  const unsigned propGrid = (unsigned)(((long long)E * 32 + 255) / 256);

  // accumulators + one-time zero of the scatter buffer (k_update re-zeroes it)
  k_zero<<<1, 32, 0, stream>>>(acc, 8);
  k_zero<<<g1(nNF), TB, 0, stream>>>(G, nNF);

  // degrees -> d_invsqrt
  k_zero<<<g1(N), TB, 0, stream>>>(dinv, N);
  k_deg<<<g1(E), TB, 0, stream>>>(dst, dinv, E);
  k_dinv<<<g1(N), TB, 0, stream>>>(dinv, N);

  // ---------------- encoder enc_dec(features) -> P -------------------------
  k_gemm<FDIM, 1><<<ggrid(N, FDIM), gblk, 0, stream>>>(features, enc_w1, enc_b1, Xa, N, FDIM);
  k_prop<<<propGrid, TB, 0, stream>>>(Xa, dinv, src, dst, G, E);
  k_update<<<g1(nNF), TB, 0, stream>>>(Xa, G, dinv, Xb, nNF);
  k_prop<<<propGrid, TB, 0, stream>>>(Xb, dinv, src, dst, G, E);
  k_update<<<g1(nNF), TB, 0, stream>>>(Xb, G, dinv, Xc, nNF);
  k_hs<<<g1(nNF), TB, 0, stream>>>(Xa, Xb, Xc, HS, nNF);
  k_gemm<HSDIM, 0><<<ggrid(N, FDIM), gblk, 0, stream>>>(HS, enc_w2, enc_b2, P, N, FDIM);

  // ---------------- VAE head: mu -> z (+KL), z -> decoder input ------------
  k_gemm<FDIM, 0><<<ggrid(N, 32), gblk, 0, stream>>>(P, rep_w, rep_b, MU, N, 32);
  k_zkl<<<g1((long long)N * 32), TB, 0, stream>>>(MU, acc + 1, (long long)N * 32);
  k_gemm<32, 0><<<ggrid(N, FDIM), gblk, 0, stream>>>(MU, rec_w, rec_b, ZN, N, FDIM);

  // ---------------- decoder enc_dec(ZN) -> x_rec (in G) --------------------
  k_gemm<FDIM, 1><<<ggrid(N, FDIM), gblk, 0, stream>>>(ZN, dec_w1, dec_b1, Xa, N, FDIM);
  k_prop<<<propGrid, TB, 0, stream>>>(Xa, dinv, src, dst, G, E);
  k_update<<<g1(nNF), TB, 0, stream>>>(Xa, G, dinv, Xb, nNF);
  k_prop<<<propGrid, TB, 0, stream>>>(Xb, dinv, src, dst, G, E);
  k_update<<<g1(nNF), TB, 0, stream>>>(Xb, G, dinv, Xc, nNF);
  k_hs<<<g1(nNF), TB, 0, stream>>>(Xa, Xb, Xc, HS, nNF);
  k_gemm<HSDIM, 0><<<ggrid(N, FDIM), gblk, 0, stream>>>(HS, dec_w2, dec_b2, G, N, FDIM);

  // ---------------- losses --------------------------------------------------
  k_sq<<<g1(nNF), TB, 0, stream>>>(G, features, acc, nNF);
  k_final<<<1, 32, 0, stream>>>(acc, (float*)d_out);
}
